// MyRNN_5557687681278
// MI455X (gfx1250) — matrix operand order repaired and checked by reading, not compiled, not run on hardware
//
#include <hip/hip_runtime.h>
#include <hip/hip_bf16.h>
#include <math.h>

#define SEQ    2048
#define BATCH  64
#define NIN    256
#define NHID   512
#define NOUT   128
#define NWG_R  16   // persistent workgroups for the recurrence (512 cols / 32)

typedef __attribute__((ext_vector_type(16))) _Float16 v16h;
typedef __attribute__((ext_vector_type(8)))  float    v8f;

union Frag16 {            // A/B operand fragment: 16 f16 = 8 VGPRs
    uint4 q[2];
    v16h  v;
};
union FragC {             // C/D accumulator fragment: 8 f32 = 8 VGPRs
    uint4 q[2];
    v8f   v;
    float f[8];
};

__device__ __forceinline__ float fast_tanh(float x) {
#if __has_builtin(__builtin_amdgcn_tanhf)
    return __builtin_amdgcn_tanhf(x);   // v_tanh_f32 (gfx1250 trans op)
#else
    return tanhf(x);
#endif
}

// ---------------------------------------------------------------------------
// f32 -> f16 conversion (weights)
// ---------------------------------------------------------------------------
__global__ void rnn_cvt_f16(const float* __restrict__ in,
                            _Float16* __restrict__ out, int n) {
    int i = blockIdx.x * blockDim.x + threadIdx.x;
    int stride = gridDim.x * blockDim.x;
    for (; i < n; i += stride) out[i] = (_Float16)in[i];
}

// ---------------------------------------------------------------------------
// zero h double-buffer + reset barrier counter (runs before recurrence)
// ---------------------------------------------------------------------------
__global__ void rnn_init(_Float16* __restrict__ hbuf, unsigned* __restrict__ bar) {
    int i = blockIdx.x * blockDim.x + threadIdx.x;
    int n = 2 * BATCH * NHID;
    for (int j = i; j < n; j += gridDim.x * blockDim.x) hbuf[j] = (_Float16)0.0f;
    if (i == 0) bar[0] = 0u;
}

// ---------------------------------------------------------------------------
// Phase 1: U[s] = x[s] @ W_ih^T + b_ih + b_hh      (one WG per timestep s)
//   Bandwidth-bound, 2048-WG parallel. unroll(2) without an occupancy cap:
//   keeps the two B panels staged as real load clauses (~230 VGPRs, under
//   the 256 boundary) while leaving room for multi-wave concurrency.
//   (Round 3: full unroll -> >256 VGPRs + MSB games. Round 4: hard cap ->
//   staging collapsed to just-in-time load->wait0->wmma.)
//   U is written fragment-major so phase 2 loads C with two b128s:
//     U[(((s*16 + chunk)*8 + wave) * 256) + lane*8 + r]
// ---------------------------------------------------------------------------
__global__ __launch_bounds__(256)
void rnn_gemm_ih(const float* __restrict__ x, const _Float16* __restrict__ Wih,
                 const float* __restrict__ b_ih, const float* __restrict__ b_hh,
                 float* __restrict__ U) {
    __shared__ _Float16 sA[BATCH * NIN];   // 32 KB
    const int s   = blockIdx.x;
    const int tid = threadIdx.x;

    // stage x[s] (f32) -> f16 LDS
    const float4* xs = (const float4*)(x + (size_t)s * BATCH * NIN);
    for (int i = tid; i < BATCH * NIN / 4; i += 256) {
        float4 f = xs[i];
        uint2 p;
        unsigned short h0 = __builtin_bit_cast(unsigned short, (_Float16)f.x);
        unsigned short h1 = __builtin_bit_cast(unsigned short, (_Float16)f.y);
        unsigned short h2 = __builtin_bit_cast(unsigned short, (_Float16)f.z);
        unsigned short h3 = __builtin_bit_cast(unsigned short, (_Float16)f.w);
        p.x = (unsigned)h0 | ((unsigned)h1 << 16);
        p.y = (unsigned)h2 | ((unsigned)h3 << 16);
        ((uint2*)sA)[i] = p;
    }
    __syncthreads();

    const int wave  = tid >> 5, lane = tid & 31;
    const int lhalf = lane & 15, hi = lane >> 4;
    const int mBase = (wave & 3) * 16;       // row tile within 64
    const int nOff  = (wave >> 2) * 16;      // col sub-tile within 32
    const int aRow  = mBase + lhalf;

    // hoist the 8 A fragments (K = 256) into registers
    Frag16 aF[8];
#pragma unroll
    for (int kc = 0; kc < 8; ++kc) {
        aF[kc].q[0] = *(const uint4*)(sA + aRow * NIN + kc * 32 + hi * 8);
        aF[kc].q[1] = *(const uint4*)(sA + aRow * NIN + kc * 32 + 16 + hi * 8);
    }

#pragma unroll 2
    for (int chunk = 0; chunk < 16; ++chunk) {        // 32 cols per chunk
        const int gcol = chunk * 32 + nOff + lhalf;
        const float bias = b_ih[gcol] + b_hh[gcol];

        // stage the whole B panel for this chunk first (16 b128 loads)
        Frag16 bF[8];
        const _Float16* brow = Wih + (size_t)gcol * NIN;
#pragma unroll
        for (int kc = 0; kc < 8; ++kc) {
            bF[kc].q[0] = *(const uint4*)(brow + kc * 32 + hi * 8);
            bF[kc].q[1] = *(const uint4*)(brow + kc * 32 + 16 + hi * 8);
        }

        FragC c;
#pragma unroll
        for (int r = 0; r < 8; ++r) c.f[r] = bias;
#pragma unroll
        for (int kc = 0; kc < 8; ++kc)
            c.v = __builtin_amdgcn_wmma_f32_16x16x32_f16(
                false, aF[kc].v, false, bF[kc].v, (short)0, c.v, false, false);

        // fragment-major U store: 2 x b128 per lane
        float* up = U + ((((size_t)s * 16 + chunk) * 8 + wave) << 8) + (lane << 3);
        ((uint4*)up)[0] = c.q[0];
        ((uint4*)up)[1] = c.q[1];
    }
}

// ---------------------------------------------------------------------------
// Phase 2: persistent recurrence, 16 WGs x 256 threads.
//   WG w owns cols [w*32, w*32+32). W_hh B-fragments (128 VGPRs) are
//   register-resident for all 2048 steps. A (h) fragments are staged through
//   an 8-slot register pipeline: slots 0..7 loaded as one clause (one L2
//   latency per step), each slot refilled with fragment kc+8 right after the
//   WMMA that consumes it. U fragments are register double-buffered. Two
//   accumulator chains halve WMMA dependency depth. h (f16, double-buffered
//   in global) is exchanged per step; atomic-counter barrier syncs the WGs.
//   Only 16 resident WGs -> no launch_bounds cap; let it keep registers.
// ---------------------------------------------------------------------------
__global__ __launch_bounds__(256)
void rnn_recur(const float* __restrict__ U, const _Float16* __restrict__ Whh,
               _Float16* __restrict__ hbuf, unsigned* __restrict__ bar) {
    const int wg  = blockIdx.x;
    const int tid = threadIdx.x;
    const int wave  = tid >> 5, lane = tid & 31;
    const int lhalf = lane & 15, hi = lane >> 4;
    const int mBase = (wave & 3) * 16;
    const int nOff  = (wave >> 2) * 16;
    const int aRow  = mBase + lhalf;
    const int gcol  = wg * 32 + nOff + lhalf;

    // register-resident B: W_hh rows [gcol], all K = 512 -> 16 fragments
    Frag16 B[16];
    {
        const _Float16* brow = Whh + (size_t)gcol * NHID;
#pragma unroll
        for (int kc = 0; kc < 16; ++kc) {
            B[kc].q[0] = *(const uint4*)(brow + kc * 32 + hi * 8);
            B[kc].q[1] = *(const uint4*)(brow + kc * 32 + 16 + hi * 8);
        }
    }

    // U fragment pointer for this (wg, wave, lane); stride 32768 floats per s
    const float* uBase = U + (((size_t)wg * 8 + wave) << 8) + (lane << 3);
    FragC uCur, uNext;
    uCur.q[0] = ((const uint4*)uBase)[0];
    uCur.q[1] = ((const uint4*)uBase)[1];

    for (int s = 0; s < SEQ; ++s) {
        const _Float16* hOld = hbuf + (size_t)(s & 1) * BATCH * NHID;
        _Float16*       hNew = hbuf + (size_t)((s + 1) & 1) * BATCH * NHID;

        // prefetch next step's U fragment (independent of the recurrence)
        {
            const int sn = (s + 1 < SEQ) ? (s + 1) : s;
            const float* un = uBase + (size_t)sn * (16 * 8 * 256);
            uNext.q[0] = ((const uint4*)un)[0];
            uNext.q[1] = ((const uint4*)un)[1];
        }

        const _Float16* aBase = hOld + (size_t)aRow * NHID;

        // stage A fragments kc = 0..7 as one load clause
        Frag16 A[8];
#pragma unroll
        for (int kc = 0; kc < 8; ++kc) {
            A[kc].q[0] = *(const uint4*)(aBase + kc * 32 + hi * 8);
            A[kc].q[1] = *(const uint4*)(aBase + kc * 32 + 16 + hi * 8);
        }

        // two independent accumulator chains; c0 starts from U (has biases)
        FragC c0, c1;
        c0.v = uCur.v;
#pragma unroll
        for (int r = 0; r < 8; ++r) c1.f[r] = 0.0f;

        // first half: consume slot kc, then refill it with fragment kc+8
#pragma unroll
        for (int kc = 0; kc < 8; ++kc) {
            if (kc & 1)
                c1.v = __builtin_amdgcn_wmma_f32_16x16x32_f16(
                    false, A[kc].v, false, B[kc].v, (short)0, c1.v, false, false);
            else
                c0.v = __builtin_amdgcn_wmma_f32_16x16x32_f16(
                    false, A[kc].v, false, B[kc].v, (short)0, c0.v, false, false);
            A[kc].q[0] = *(const uint4*)(aBase + (kc + 8) * 32 + hi * 8);
            A[kc].q[1] = *(const uint4*)(aBase + (kc + 8) * 32 + 16 + hi * 8);
        }
        // second half
#pragma unroll
        for (int kc = 8; kc < 16; ++kc) {
            if (kc & 1)
                c1.v = __builtin_amdgcn_wmma_f32_16x16x32_f16(
                    false, A[kc - 8].v, false, B[kc].v, (short)0, c1.v, false, false);
            else
                c0.v = __builtin_amdgcn_wmma_f32_16x16x32_f16(
                    false, A[kc - 8].v, false, B[kc].v, (short)0, c0.v, false, false);
        }
        v8f av = c0.v + c1.v;

        // tanh (hardware v_tanh_f32 when available) + store new h as f16
        _Float16* hrow = hNew + (size_t)(mBase + hi * 8) * NHID + gcol;
#pragma unroll
        for (int r = 0; r < 8; ++r)
            hrow[(size_t)r * NHID] = (_Float16)fast_tanh(av[r]);

        // ---- device-wide barrier across the 16 persistent WGs ----
        __threadfence();            // release our h stores to L2
        __syncthreads();
        if (tid == 0) {
            atomicAdd(bar, 1u);
            const unsigned goal = (unsigned)(s + 1) * NWG_R;
            while (atomicAdd(bar, 0u) < goal) __builtin_amdgcn_s_sleep(1);
        }
        __syncthreads();
        __threadfence();            // acquire: invalidate stale h lines

        uCur = uNext;
    }
}

// ---------------------------------------------------------------------------
// Phase 3: out = h_final @ W_ho^T + b_ho   (64x128, trivial)
//   h_final lives in hbuf buffer 0 (written by step s=2047).
// ---------------------------------------------------------------------------
__global__ void rnn_out(const _Float16* __restrict__ hfin,
                        const float* __restrict__ Who,
                        const float* __restrict__ b_ho,
                        float* __restrict__ out) {
    int idx = blockIdx.x * blockDim.x + threadIdx.x;
    if (idx >= BATCH * NOUT) return;
    const int bt = idx / NOUT, o = idx % NOUT;
    const _Float16* hr = hfin + (size_t)bt * NHID;
    const float*    wr = Who + (size_t)o * NHID;
    float sum = b_ho[o];
    for (int k = 0; k < NHID; ++k) sum += (float)hr[k] * wr[k];
    out[idx] = sum;
}

// ---------------------------------------------------------------------------
extern "C" void kernel_launch(void* const* d_in, const int* in_sizes, int n_in,
                              void* d_out, int out_size, void* d_ws, size_t ws_size,
                              hipStream_t stream) {
    const float* x   = (const float*)d_in[0];
    const float* Wih = (const float*)d_in[1];
    const float* bih = (const float*)d_in[2];
    const float* Whh = (const float*)d_in[3];
    const float* bhh = (const float*)d_in[4];
    const float* Who = (const float*)d_in[5];
    const float* bho = (const float*)d_in[6];

    char* ws = (char*)d_ws;
    size_t off = 0;
    float* U = (float*)(ws + off);            off += (size_t)SEQ * BATCH * NHID * 4; // 256 MiB
    _Float16* Wih_h = (_Float16*)(ws + off);  off += (size_t)NHID * NIN * 2;
    _Float16* Whh_h = (_Float16*)(ws + off);  off += (size_t)NHID * NHID * 2;
    _Float16* hbuf  = (_Float16*)(ws + off);  off += (size_t)2 * BATCH * NHID * 2;
    unsigned* bar   = (unsigned*)(ws + off);

    rnn_cvt_f16<<<256, 256, 0, stream>>>(Wih, Wih_h, NHID * NIN);
    rnn_cvt_f16<<<512, 256, 0, stream>>>(Whh, Whh_h, NHID * NHID);
    rnn_init<<<64, 256, 0, stream>>>(hbuf, bar);

    rnn_gemm_ih<<<SEQ, 256, 0, stream>>>(x, Wih_h, bih, bhh, U);
    rnn_recur<<<NWG_R, 256, 0, stream>>>(U, Whh_h, hbuf, bar);
    // final h is in buffer 0 (step 2047 writes buffer (2047+1)&1 == 0)
    rnn_out<<<(BATCH * NOUT + 255) / 256, 256, 0, stream>>>(hbuf, Who, bho,
                                                            (float*)d_out);
}
